// SloughGPTAttention_7937099563414
// MI455X (gfx1250) — matrix operand order repaired and checked by reading, not compiled, not run on hardware
//
#include <hip/hip_runtime.h>
#include <hip/hip_bf16.h>
#include <stdint.h>

// ---------------------------------------------------------------------------
// CDNA5 (gfx1250) bf16 WMMA building blocks.
// v_wmma_f32_16x16x32_bf16: D(16x16 f32) = A(16x32 bf16) * B(32x16 bf16) + C
// Fragment layouts used here (wave32; C/D per CDNA5 ISA §7.12.2):
//  A: lane m=l&15 holds row m; K contiguous, this code's order (ISA table differs): (hi?16:0)+0..15
//  B: lane n=l&15 holds col n; K contiguous, this code's order (ISA table differs): (hi?16:0)+0..15
//  C/D: lane col n=l&15; vgpr v -> row m = v + (hi?8:0)
// ---------------------------------------------------------------------------

typedef __attribute__((ext_vector_type(16))) __bf16 bfx16;
typedef __attribute__((ext_vector_type(8)))  __bf16 bfx8;
typedef __attribute__((ext_vector_type(4)))  __bf16 bfx4;
typedef __attribute__((ext_vector_type(8)))  float  fx8;
typedef __attribute__((ext_vector_type(4)))  float  fx4;

union BFrag { bfx16 v; bfx8 h[2]; };

__device__ __forceinline__ fx8 wmma_bf16(const BFrag& a, const BFrag& b, fx8 c) {
  return __builtin_amdgcn_wmma_f32_16x16x32_bf16(
      /*neg_a=*/false, a.v, /*neg_b=*/false, b.v,
      /*c_mod=*/(short)0, c, /*reuse_a=*/false, /*reuse_b=*/false);
}

// A-matrix fragment from row-major [M x K] source, tile origin = base.
__device__ __forceinline__ BFrag load_a_frag(const __bf16* base, int ld, int lane) {
  const int m  = lane & 15;
  const int hi = lane >> 4;
  const __bf16* p = base + (size_t)m * ld + (hi << 4);
  BFrag f;
  f.h[0] = *(const bfx8*)(p);        // K = (hi?16:0) + 0..7
  f.h[1] = *(const bfx8*)(p + 8);    // K = (hi?16:0) + 8..15
  return f;
}

// B-matrix fragment where source is row-major [N x K] (i.e. B columns are
// contiguous in memory) -- exactly the NT-GEMM weight layout.
__device__ __forceinline__ BFrag load_b_frag(const __bf16* base, int ld, int lane) {
  const int n  = lane & 15;
  const int hi = lane >> 4;
  const __bf16* p = base + (size_t)n * ld + (hi << 4);
  BFrag f;
  f.h[0] = *(const bfx8*)(p);        // K = (hi?16:0) + 0..7
  f.h[1] = *(const bfx8*)(p + 8);    // K = (hi?16:0) + 8..15
  return f;
}

// ---------------------------------------------------------------------------
// Elementwise f32 -> bf16 conversion, 4-wide (b128 load, b64 store).
// All tensor sizes here are multiples of 4.
// ---------------------------------------------------------------------------
__global__ void f32_to_bf16_kernel(const float* __restrict__ in,
                                   __bf16* __restrict__ out, size_t n4) {
  const size_t i = (size_t)blockIdx.x * blockDim.x + threadIdx.x;
  if (i < n4) {
    const fx4 v = ((const fx4*)in)[i];
    bfx4 o;
#pragma unroll
    for (int e = 0; e < 4; e++) o[e] = (__bf16)v[e];
    ((bfx4*)out)[i] = o;
  }
}

// ---------------------------------------------------------------------------
// NT GEMM: C[m][n] = sum_k A[m][k] * B[n][k]
// A: M x K bf16 row-major. B: N x K bf16 row-major.
// Block = 8 waves (256 thr) arranged 4x2 -> 128x128 tile; wave tile 32x64.
// L2 (192MB) holds the whole working set, so direct global loads are fine.
// ---------------------------------------------------------------------------
__global__ __launch_bounds__(256)
void gemm_nt_bf16(const __bf16* __restrict__ A, const __bf16* __restrict__ B,
                  float* __restrict__ Cf, __bf16* __restrict__ Cb,
                  int M, int N, int K) {
  const int lane = threadIdx.x & 31;
  const int w    = threadIdx.x >> 5;
  const int row0 = blockIdx.y * 128 + (w >> 1) * 32;
  const int col0 = blockIdx.x * 128 + (w & 1) * 64;

  fx8 acc[2][4] = {};
  for (int k0 = 0; k0 < K; k0 += 32) {
    BFrag a[2], b[4];
    a[0] = load_a_frag(A + (size_t)row0 * K + k0, K, lane);
    a[1] = load_a_frag(A + (size_t)(row0 + 16) * K + k0, K, lane);
#pragma unroll
    for (int j = 0; j < 4; j++)
      b[j] = load_b_frag(B + (size_t)(col0 + 16 * j) * K + k0, K, lane);
#pragma unroll
    for (int i = 0; i < 2; i++)
#pragma unroll
      for (int j = 0; j < 4; j++)
        acc[i][j] = wmma_bf16(a[i], b[j], acc[i][j]);
  }

  const int n = lane & 15, hi = lane >> 4;
#pragma unroll
  for (int i = 0; i < 2; i++)
#pragma unroll
    for (int j = 0; j < 4; j++) {
      const int r = row0 + 16 * i + hi * 8;
      const int c = col0 + 16 * j + n;
#pragma unroll
      for (int v = 0; v < 8; v++) {
        const size_t off = (size_t)(r + v) * N + c;
        const float val = acc[i][j][v];
        if (Cf) Cf[off] = val;
        else    Cb[off] = (__bf16)val;
      }
    }
}

// ---------------------------------------------------------------------------
// In-place RoPE on bf16 projections laid out [rows][heads*128].
// Thread handles one (row, head, d) pair with d in 0..63:
//   y[d]    = x[d]*cos - x[d+64]*sin
//   y[d+64] = x[d+64]*cos + x[d]*sin,  angle = t * 10000^(-d/64)
// ---------------------------------------------------------------------------
__global__ void rope_inplace_kernel(__bf16* __restrict__ Q, int rows, int heads) {
  const int idx = blockIdx.x * blockDim.x + threadIdx.x;
  const int total = rows * heads * 64;
  if (idx >= total) return;
  const int d  = idx & 63;
  const int hh = (idx >> 6) % heads;
  const int r  = idx / (64 * heads);
  const int t  = r & 2047;  // r % T
  // inv_freq = 10000^(-2d/128) = exp(-d * ln(10000)/64)
  const float inv = __expf(-(float)d * (9.210340371976184f / 64.0f));
  const float ang = (float)t * inv;
  const float c = __cosf(ang), s = __sinf(ang);
  __bf16* p = Q + (size_t)r * (heads * 128) + hh * 128 + d;
  const float x0 = (float)p[0], x1 = (float)p[64];
  p[0]  = (__bf16)(x0 * c - x1 * s);
  p[64] = (__bf16)(x1 * c + x0 * s);
}

// ---------------------------------------------------------------------------
// Transpose V: Vb [(b,t) x (kv*128)] bf16 -> Vt [(b,kv,d) x t] bf16
// so that P@V B-fragments become contiguous loads (lane = d column,
// K(=s) contiguous). 32x32 tiles.
// Tile fill uses the CDNA5 async global->LDS path (GLOBAL_LOAD_ASYNC_TO_LDS,
// tracked by ASYNCcnt) instead of VGPR round-trips; LDS offset is the low
// 32 bits of the generic pointer (LDS aperture maps addr[31:0] -> LDS).
// ---------------------------------------------------------------------------
__global__ __launch_bounds__(256)
void transpose_v_kernel(const __bf16* __restrict__ Vb, __bf16* __restrict__ Vt) {
  __shared__ __align__(8) __bf16 tile[32][34];   // pad 2 -> rows stay 4B-aligned
  const int c0 = blockIdx.x * 32;          // col in Vb (kv*128+d), 0..511
  const int r0 = blockIdx.y * 32;          // row in Vb (b*2048+t), 0..4095
  const int tp = (threadIdx.x & 15) * 2;   // bf16 pair within a row
  const int tr = threadIdx.x >> 4;         // 0..15

#pragma unroll
  for (int i = 0; i < 2; i++) {
    const int r = tr + i * 16;
    const uint32_t lds_off =
        (uint32_t)(uintptr_t)&tile[r][tp];
    const uint64_t gaddr =
        (uint64_t)(uintptr_t)(Vb + (size_t)(r0 + r) * 512 + c0 + tp);
    asm volatile("global_load_async_to_lds_b32 %0, %1, off"
                 :: "v"(lds_off), "v"(gaddr)
                 : "memory");
  }
  asm volatile("s_wait_asynccnt 0x0" ::: "memory");
  __syncthreads();

  const int tx = threadIdx.x & 31;
  const int ty = threadIdx.x >> 5;         // 0..7
#pragma unroll
  for (int i = 0; i < 4; i++) {
    const int rr  = ty + i * 8;            // local col of Vb
    const int col = c0 + rr;               // kv*128 + d
    const int kv  = col >> 7, d = col & 127;
    const int r   = r0 + tx;               // b*2048 + t
    const int b   = r >> 11, t = r & 2047;
    Vt[(((size_t)b * 4 + kv) * 128 + d) * 2048 + t] = tile[tx][rr];
  }
}

// ---------------------------------------------------------------------------
// Flash attention (causal, GQA). Grid (T/64, B*H), 128 threads = 4 waves.
// Each wave: 16 query rows, full D=128 in registers. KV tiles of 32.
//   S  = Q(16x128) @ K^T          -> 8 WMMA (2 n-tiles x 4 k-steps)
//   O += P(16x32) @ V             -> 8 WMMA (8 d-tiles), V from Vt (d-major)
// Online softmax with half-wave butterfly reductions (wave32 C layout:
// rows 0-7 in lanes 0-15, rows 8-15 in lanes 16-31).
// ---------------------------------------------------------------------------
__global__ __launch_bounds__(128)
void attn_kernel(const __bf16* __restrict__ Qb, const __bf16* __restrict__ Kb,
                 const __bf16* __restrict__ Vt, __bf16* __restrict__ Ao) {
  __shared__ __align__(16) __bf16 pt[4][16][40];   // per-wave P bounce buffer

  const int lane = threadIdx.x & 31;
  const int w    = threadIdx.x >> 5;
  const int n    = lane & 15;
  const int hi   = lane >> 4;

  const int bh = blockIdx.y;           // 0..31
  const int b  = bh >> 4, h = bh & 15;
  const int kv = h >> 2;               // GQA: head h -> kv head h/4
  const int q0 = blockIdx.x * 64 + w * 16;

  const __bf16* Qp = Qb + ((size_t)b * 2048 + q0) * 2048 + h * 128;
  const __bf16* Kp = Kb + ((size_t)b * 2048) * 512 + kv * 128;
  const __bf16* Vp = Vt + (((size_t)b * 4 + kv) * 128) * 2048;

  // Preload Q fragments (4 k-chunks of 32 over D=128)
  BFrag qf[4];
#pragma unroll
  for (int kk = 0; kk < 4; kk++) qf[kk] = load_a_frag(Qp + kk * 32, 2048, lane);

  fx8 oacc[8] = {};
  float mrun[8], lrun[8];
#pragma unroll
  for (int v = 0; v < 8; v++) { mrun[v] = -1e30f; lrun[v] = 0.0f; }

  const float scale = 0.08838834764831843f;  // 1/sqrt(128)
  const int qrow_base = q0 + hi * 8;

  for (int s0 = 0; s0 < q0 + 16; s0 += 32) {
    // ---- S = Q @ K^T ----
    fx8 st[2] = {};
#pragma unroll
    for (int jt = 0; jt < 2; jt++)
#pragma unroll
      for (int kk = 0; kk < 4; kk++) {
        BFrag kf = load_b_frag(Kp + (size_t)(s0 + jt * 16) * 512 + kk * 32,
                               512, lane);
        st[jt] = wmma_bf16(qf[kk], kf, st[jt]);
      }

    // ---- scale + causal mask + row max ----
    float rmax[8];
#pragma unroll
    for (int v = 0; v < 8; v++) rmax[v] = -1e30f;
#pragma unroll
    for (int jt = 0; jt < 2; jt++) {
      const int sg = s0 + jt * 16 + n;
#pragma unroll
      for (int v = 0; v < 8; v++) {
        float val = st[jt][v] * scale;
        if (sg > qrow_base + v) val = -1e30f;
        st[jt][v] = val;
        rmax[v] = fmaxf(rmax[v], val);
      }
    }
#pragma unroll
    for (int msk = 1; msk < 16; msk <<= 1)
#pragma unroll
      for (int v = 0; v < 8; v++)
        rmax[v] = fmaxf(rmax[v], __shfl_xor(rmax[v], msk, 32));

    // ---- online softmax update ----
    float alpha[8], rsum[8];
#pragma unroll
    for (int v = 0; v < 8; v++) {
      const float mn = fmaxf(mrun[v], rmax[v]);
      alpha[v] = __expf(mrun[v] - mn);
      mrun[v] = mn;
      rsum[v] = 0.0f;
    }
#pragma unroll
    for (int jt = 0; jt < 2; jt++)
#pragma unroll
      for (int v = 0; v < 8; v++) {
        const float pv = __expf(st[jt][v] - mrun[v]);
        st[jt][v] = pv;
        rsum[v] += pv;
      }
#pragma unroll
    for (int msk = 1; msk < 16; msk <<= 1)
#pragma unroll
      for (int v = 0; v < 8; v++) rsum[v] += __shfl_xor(rsum[v], msk, 32);
#pragma unroll
    for (int v = 0; v < 8; v++) lrun[v] = lrun[v] * alpha[v] + rsum[v];
#pragma unroll
    for (int j = 0; j < 8; j++)
#pragma unroll
      for (int v = 0; v < 8; v++) oacc[j][v] *= alpha[v];

    // ---- P: C-layout -> A-fragment via per-wave LDS (in-order DS) ----
#pragma unroll
    for (int jt = 0; jt < 2; jt++)
#pragma unroll
      for (int v = 0; v < 8; v++)
        pt[w][hi * 8 + v][jt * 16 + n] = (__bf16)st[jt][v];

    BFrag pf;
    {
      const __bf16* pp = &pt[w][n][hi * 16];
      pf.h[0] = *(const bfx8*)(pp);
      pf.h[1] = *(const bfx8*)(pp + 8);
    }

    // ---- O += P @ V (V from Vt: lane = d column, s contiguous) ----
#pragma unroll
    for (int j = 0; j < 8; j++) {
      BFrag vf = load_b_frag(Vp + (size_t)(j * 16) * 2048 + s0, 2048, lane);
      oacc[j] = wmma_bf16(pf, vf, oacc[j]);
    }
  }

  // ---- epilogue: normalize, store bf16 attn-out [(b,t) x (h*128)] ----
#pragma unroll
  for (int v = 0; v < 8; v++) {
    const float inv_l = 1.0f / lrun[v];
    const int m = hi * 8 + v;
#pragma unroll
    for (int j = 0; j < 8; j++) {
      Ao[((size_t)b * 2048 + q0 + m) * 2048 + h * 128 + j * 16 + n] =
          (__bf16)(oacc[j][v] * inv_l);
    }
  }
}

// ---------------------------------------------------------------------------
// Host-side launcher
// ---------------------------------------------------------------------------
extern "C" void kernel_launch(void* const* d_in, const int* in_sizes, int n_in,
                              void* d_out, int out_size, void* d_ws, size_t ws_size,
                              hipStream_t stream) {
  (void)n_in; (void)out_size; (void)ws_size;
  const float* x  = (const float*)d_in[0];
  const float* Wq = (const float*)d_in[1];
  const float* Wk = (const float*)d_in[2];
  const float* Wv = (const float*)d_in[3];
  const float* Wo = (const float*)d_in[4];
  float* out = (float*)d_out;

  const int T = 2048, C = 2048, H = 16, KV = 4;
  const int M = 2 * T;  // B*T = 4096
  const size_t nx  = (size_t)in_sizes[0];   // 8388608
  const size_t nwq = (size_t)in_sizes[1];   // 4194304
  const size_t nwk = (size_t)in_sizes[2];   // 1048576
  const size_t nwv = (size_t)in_sizes[3];
  const size_t nwo = (size_t)in_sizes[4];

  char* p = (char*)d_ws;
  auto alloc = [&](size_t bytes) -> char* {
    char* r = p;
    p += (bytes + 255) & ~(size_t)255;
    return r;
  };
  __bf16* xb  = (__bf16*)alloc(nx * 2);
  __bf16* wqb = (__bf16*)alloc(nwq * 2);
  __bf16* wkb = (__bf16*)alloc(nwk * 2);
  __bf16* wvb = (__bf16*)alloc(nwv * 2);
  __bf16* wob = (__bf16*)alloc(nwo * 2);
  __bf16* Qb  = (__bf16*)alloc((size_t)M * C * 2);
  __bf16* Kb  = (__bf16*)alloc((size_t)M * 512 * 2);
  __bf16* Vb  = (__bf16*)alloc((size_t)M * 512 * 2);
  __bf16* Vt  = (__bf16*)alloc((size_t)M * 512 * 2);
  __bf16* Ao  = (__bf16*)alloc((size_t)M * C * 2);

  // 1) convert inputs to bf16 (4 elems/thread)
  f32_to_bf16_kernel<<<(unsigned)((nx / 4  + 255) / 256), 256, 0, stream>>>(x,  xb,  nx / 4);
  f32_to_bf16_kernel<<<(unsigned)((nwq / 4 + 255) / 256), 256, 0, stream>>>(Wq, wqb, nwq / 4);
  f32_to_bf16_kernel<<<(unsigned)((nwk / 4 + 255) / 256), 256, 0, stream>>>(Wk, wkb, nwk / 4);
  f32_to_bf16_kernel<<<(unsigned)((nwv / 4 + 255) / 256), 256, 0, stream>>>(Wv, wvb, nwv / 4);
  f32_to_bf16_kernel<<<(unsigned)((nwo / 4 + 255) / 256), 256, 0, stream>>>(Wo, wob, nwo / 4);

  // 2) QKV projections (NT GEMMs, WMMA, bf16 out)
  gemm_nt_bf16<<<dim3(C / 128, M / 128), 256, 0, stream>>>(xb, wqb, nullptr, Qb, M, C, C);
  gemm_nt_bf16<<<dim3(512 / 128, M / 128), 256, 0, stream>>>(xb, wkb, nullptr, Kb, M, 512, C);
  gemm_nt_bf16<<<dim3(512 / 128, M / 128), 256, 0, stream>>>(xb, wvb, nullptr, Vb, M, 512, C);

  // 3) RoPE in-place on Q and K
  rope_inplace_kernel<<<(M * H  * 64 + 255) / 256, 256, 0, stream>>>(Qb, M, H);
  rope_inplace_kernel<<<(M * KV * 64 + 255) / 256, 256, 0, stream>>>(Kb, M, KV);

  // 4) V -> d-major layout for WMMA B-fragments in P@V (async global->LDS)
  transpose_v_kernel<<<dim3(512 / 32, M / 32), 256, 0, stream>>>(Vb, Vt);

  // 5) causal GQA flash attention
  attn_kernel<<<dim3(T / 64, 2 * H), 128, 0, stream>>>(Qb, Kb, Vt, Ao);

  // 6) output projection (f32 out)
  gemm_nt_bf16<<<dim3(C / 128, M / 128), 256, 0, stream>>>(Ao, wob, out, nullptr, M, C, C);
}
